// Decoder_14757507629300
// MI455X (gfx1250) — compile-verified, matrix operand order checked by reading, not run on hardware
//
#include <hip/hip_runtime.h>
#include <hip/hip_bf16.h>

// ---------------------------------------------------------------------------
// MI455X (gfx1250) attention-LSTM decoder.
// B=256, T_IN=512, T_OUT=256 (255 decoded steps), U=256, V=6.
//
// Strategy (see analysis): everything streamed per-step (keys, memory) is
// converted to bf16 once so the 134 MB working set is L2-resident; all GEMMs
// use v_wmma_f32_16x16x32_bf16 with f32 accumulation; the one-hot input GEMM
// is folded into a [6,1024] bias table; weights are pre-transposed so WMMA
// B-fragments load as two contiguous b128 chunks per lane.
// ---------------------------------------------------------------------------

typedef __bf16 bf16_t;
typedef __attribute__((ext_vector_type(16))) __bf16 v16bf;
typedef __attribute__((ext_vector_type(8)))  __bf16 v8bf;
typedef __attribute__((ext_vector_type(8)))  float  v8f;

#define NB    256          // batch
#define TIN   512          // memory length
#define TOUT  256          // ids length (decode TOUT-1 steps)
#define NU    256          // units
#define NV    6            // vocab

__device__ __forceinline__ bf16_t f2bf(float f) {
  unsigned x = __builtin_bit_cast(unsigned, f);
  unsigned r = (x + 0x7FFFu + ((x >> 16) & 1u)) >> 16;   // RNE
  unsigned short h = (unsigned short)r;
  return __builtin_bit_cast(bf16_t, h);
}
__device__ __forceinline__ float bf2f(bf16_t b) {
  unsigned short h = __builtin_bit_cast(unsigned short, b);
  unsigned x = ((unsigned)h) << 16;
  return __builtin_bit_cast(float, x);
}
__device__ __forceinline__ float sigm(float x) { return 1.0f / (1.0f + __expf(-x)); }

// 16-bit A-matrix 16x32 layout (ISA 7.12.2): lane half h, row r = lane&15,
// elements are K = {h*8..h*8+7} and {16+h*8..16+h*8+7}: two contiguous runs
// of 8 bf16 (16B each).  B matrices are stored pre-transposed (column-major,
// Wt[n*K + k]) so the identical loader serves both operands.
__device__ __forceinline__ v16bf load_frag(const bf16_t* __restrict__ base,
                                           int ld, int r, int kb, int half) {
  const bf16_t* p = base + (size_t)r * ld + kb + half * 8;
  v8bf lo = *(const v8bf*)(p);
  v8bf hi = *(const v8bf*)(p + 16);
  v16bf o;
#pragma unroll
  for (int i = 0; i < 8; ++i) { o[i] = lo[i]; o[i + 8] = hi[i]; }
  return o;
}

__device__ __forceinline__ v8f wmma_bf16(v16bf a, v16bf b, v8f c) {
  return __builtin_amdgcn_wmma_f32_16x16x32_bf16(false, a, false, b,
                                                 (short)0, c, false, false);
}

// ---------------------------------------------------------------------------
// Setup kernels
// ---------------------------------------------------------------------------

__global__ void k_cvt_mem(const float* __restrict__ src, bf16_t* __restrict__ dst, int n) {
  int i = blockIdx.x * 256 + threadIdx.x;
  if (i < n) dst[i] = f2bf(src[i]);
}

// Pack all weights to bf16, column-major; build zb[id][col] = Wi[id,col] + b[col].
__global__ void k_pack(const float* __restrict__ Wi, const float* __restrict__ Wh,
                       const float* __restrict__ bias,
                       const float* __restrict__ Wq, const float* __restrict__ Wa,
                       const float* __restrict__ Wm,
                       bf16_t* __restrict__ Wzn_t, bf16_t* __restrict__ Wq_t,
                       bf16_t* __restrict__ Wa_t, bf16_t* __restrict__ Wm_t,
                       float* __restrict__ zb) {
  int i = blockIdx.x * 256 + threadIdx.x;
  if (i < 524288) {   // Wzn_t[n*512+k]: vertical stack [Wi rows 6..261 ; Wh]
    int n = i >> 9, k = i & 511;
    float v = (k < 256) ? Wi[(size_t)(6 + k) * 1024 + n] : Wh[(size_t)(k - 256) * 1024 + n];
    Wzn_t[i] = f2bf(v);
    return;
  }
  i -= 524288;
  if (i < 65536) { int n = i >> 8, k = i & 255; Wq_t[i] = f2bf(Wq[(size_t)k * 256 + n]); return; }
  i -= 65536;
  if (i < 131072) { int n = i >> 9, k = i & 511; Wa_t[i] = f2bf(Wa[(size_t)k * 256 + n]); return; }
  i -= 131072;
  if (i < 65536) { int n = i >> 8, k = i & 255; Wm_t[i] = f2bf(Wm[(size_t)k * 256 + n]); return; }
  i -= 65536;
  if (i < 6144) { int vr = i >> 10, c = i & 1023; zb[i] = Wi[(size_t)vr * 1024 + c] + bias[c]; }
}

// Xz = [attn0=0 | bf16(enc_h)], c_state = enc_c.  One thread per (b,u).
__global__ void k_init(const float* __restrict__ enc_h, const float* __restrict__ enc_c,
                       bf16_t* __restrict__ Xz, float* __restrict__ c_st) {
  int i = blockIdx.x * 256 + threadIdx.x;       // 65536 threads
  int b = i >> 8, u = i & 255;
  Xz[(size_t)b * 512 + u]       = f2bf(0.0f);
  Xz[(size_t)b * 512 + 256 + u] = f2bf(enc_h[i]);
  c_st[i] = enc_c[i];
}

// keys = bf16( memory @ Wm ), M = B*TIN = 131072 rows.  8 waves/block,
// each wave owns 2 column tiles; K = 256 -> 8 WMMA per tile.
__global__ __launch_bounds__(256) void k_keys(const bf16_t* __restrict__ memb,
                                              const bf16_t* __restrict__ Wm_t,
                                              bf16_t* __restrict__ keys) {
  int tid = threadIdx.x, lane = tid & 31, w = tid >> 5;
  int half = lane >> 4, r = lane & 15;
  size_t m0 = (size_t)blockIdx.x * 16;
  const bf16_t* A = memb + m0 * 256;
  int n0 = (2 * w) * 16, n1 = n0 + 16;
  v8f acc0 = {}, acc1 = {};
  for (int kb = 0; kb < 256; kb += 32) {
    v16bf a  = load_frag(A, 256, r, kb, half);
    v16bf b0 = load_frag(Wm_t + (size_t)n0 * 256, 256, r, kb, half);
    v16bf b1 = load_frag(Wm_t + (size_t)n1 * 256, 256, r, kb, half);
    acc0 = wmma_bf16(a, b0, acc0);
    acc1 = wmma_bf16(a, b1, acc1);
  }
#pragma unroll
  for (int j = 0; j < 8; ++j) {
    int m = half * 8 + j;
    keys[(m0 + m) * 256 + n0 + r] = f2bf(acc0[j]);
    keys[(m0 + m) * 256 + n1 + r] = f2bf(acc1[j]);
  }
}

// ---------------------------------------------------------------------------
// Per-step kernels
// ---------------------------------------------------------------------------

// K1: z = [attn|h] @ Wzn + zb[id]; LSTM gates; write h_new(bf16)->Xa[:,0:256],
// c_state in place.  One wave per (16-row, 16-col) h tile; 4 gate columns,
// K=512 -> 64 WMMA per wave.  Each block touches only its own rows/cols.
__global__ __launch_bounds__(32) void k_lstm(const bf16_t* __restrict__ Xz,
                                             const bf16_t* __restrict__ Wzn_t,
                                             const float* __restrict__ zb,
                                             const int* __restrict__ ids,
                                             float* __restrict__ c_st,
                                             bf16_t* __restrict__ Xa, int t) {
  int lane = threadIdx.x;
  int half = lane >> 4, r = lane & 15;
  int bi0 = blockIdx.x * 16, u0 = blockIdx.y * 16;
  const bf16_t* A = Xz + (size_t)bi0 * 512;
  v8f acc[4] = {};
  for (int kb = 0; kb < 512; kb += 32) {
    v16bf a = load_frag(A, 512, r, kb, half);
#pragma unroll
    for (int g = 0; g < 4; ++g) {
      v16bf bw = load_frag(Wzn_t + (size_t)(g * 256 + u0) * 512, 512, r, kb, half);
      acc[g] = wmma_bf16(a, bw, acc[g]);
    }
  }
  int u = u0 + r;
#pragma unroll
  for (int j = 0; j < 8; ++j) {
    int m = half * 8 + j;
    int b = bi0 + m;
    int id = ids[(size_t)b * TOUT + t];
    const float* zr = zb + (size_t)id * 1024;
    float zi = acc[0][j] + zr[u];
    float zf = acc[1][j] + zr[256 + u];
    float zg = acc[2][j] + zr[512 + u];
    float zo = acc[3][j] + zr[768 + u];
    float c_old = c_st[(size_t)b * 256 + u];
    float cn = sigm(zf) * c_old + sigm(zi) * tanhf(zg);
    float hn = sigm(zo) * tanhf(cn);
    c_st[(size_t)b * 256 + u] = cn;
    Xa[(size_t)b * 512 + u] = f2bf(hn);
  }
}

// K2: pq = h_new @ Wq (WMMA, LDS); scores s[b,t] = sum_u v[u]*tanh(keys+pq);
// softmax over t; ctx = align @ memory (bf16 streams, L2-resident);
// ctx(bf16) -> Xa[:,256:512].  16 batch rows per block, 8 waves.
__global__ __launch_bounds__(256) void k_attn(bf16_t* __restrict__ Xa,
                                              const bf16_t* __restrict__ Wq_t,
                                              const bf16_t* __restrict__ keys,
                                              const bf16_t* __restrict__ memb,
                                              const float* __restrict__ vvec) {
  __shared__ float pq_s[16][256];
  __shared__ float s_s[16][512];
  __shared__ float inv_s[16];
  int tid = threadIdx.x, lane = tid & 31, w = tid >> 5;
  int half = lane >> 4, r = lane & 15;
  int bi0 = blockIdx.x * 16;

  // Phase A: pq tiles.  A = h_new rows (Xa cols 0..255, ld=512).
  {
    const bf16_t* A = Xa + (size_t)bi0 * 512;
    int n0 = (2 * w) * 16, n1 = n0 + 16;
    v8f acc0 = {}, acc1 = {};
    for (int kb = 0; kb < 256; kb += 32) {
      v16bf a  = load_frag(A, 512, r, kb, half);
      v16bf b0 = load_frag(Wq_t + (size_t)n0 * 256, 256, r, kb, half);
      v16bf b1 = load_frag(Wq_t + (size_t)n1 * 256, 256, r, kb, half);
      acc0 = wmma_bf16(a, b0, acc0);
      acc1 = wmma_bf16(a, b1, acc1);
    }
#pragma unroll
    for (int j = 0; j < 8; ++j) {
      int m = half * 8 + j;
      pq_s[m][n0 + r] = acc0[j];
      pq_s[m][n1 + r] = acc1[j];
    }
  }
  __syncthreads();

  // Phase B: scores. Each wave handles (bi,t) pairs; 32 lanes x 8 u each.
  float vv[8];
#pragma unroll
  for (int i = 0; i < 8; ++i) vv[i] = vvec[lane * 8 + i];
  for (int idx = w; idx < 16 * TIN; idx += 8) {
    int bi = idx & 15, tt = idx >> 4;
    int b = bi0 + bi;
    const bf16_t* krow = keys + ((size_t)b * TIN + tt) * 256 + lane * 8;
    if (tt + 8 < TIN)
      __builtin_prefetch(krow + 8 * 256, 0, 1);   // global_prefetch_b8
    v8bf kv = *(const v8bf*)krow;
    float part = 0.0f;
#pragma unroll
    for (int i = 0; i < 8; ++i)
      part += vv[i] * tanhf(bf2f(kv[i]) + pq_s[bi][lane * 8 + i]);
#pragma unroll
    for (int off = 16; off; off >>= 1) part += __shfl_xor(part, off, 32);
    if (lane == 0) s_s[bi][tt] = part;
  }
  __syncthreads();

  // Phase C: softmax normalization per row (2 rows per wave).
  for (int rr = 0; rr < 2; ++rr) {
    int row = 2 * w + rr;
    float vals[16];
    float mx = -3.4e38f;
#pragma unroll
    for (int j = 0; j < 16; ++j) { vals[j] = s_s[row][lane + 32 * j]; mx = fmaxf(mx, vals[j]); }
#pragma unroll
    for (int off = 16; off; off >>= 1) mx = fmaxf(mx, __shfl_xor(mx, off, 32));
    float sum = 0.0f;
#pragma unroll
    for (int j = 0; j < 16; ++j) {
      float e = __expf(vals[j] - mx);
      s_s[row][lane + 32 * j] = e;
      sum += e;
    }
#pragma unroll
    for (int off = 16; off; off >>= 1) sum += __shfl_xor(sum, off, 32);
    if (lane == 0) inv_s[row] = 1.0f / sum;
  }
  __syncthreads();

  // Phase D: ctx = align @ memory, same row ownership as phase C.
  for (int rr = 0; rr < 2; ++rr) {
    int row = 2 * w + rr;
    int b = bi0 + row;
    float ctx[8] = {0.f, 0.f, 0.f, 0.f, 0.f, 0.f, 0.f, 0.f};
    const bf16_t* mrow = memb + (size_t)b * TIN * 256 + lane * 8;
    for (int tt = 0; tt < TIN; ++tt) {
      float wt = s_s[row][tt];
      v8bf mv = *(const v8bf*)(mrow + (size_t)tt * 256);
#pragma unroll
      for (int i = 0; i < 8; ++i) ctx[i] += wt * bf2f(mv[i]);
    }
    float inv = inv_s[row];
    v8bf ov;
#pragma unroll
    for (int i = 0; i < 8; ++i) ov[i] = f2bf(ctx[i] * inv);
    *(v8bf*)(Xa + (size_t)b * 512 + 256 + lane * 8) = ov;
  }
}

// K3: attn_new = [h_new|ctx] @ Wa (WMMA, K=512); attn_new(bf16)->Xz[:,0:256],
// copy h_new -> Xz[:,256:512] for the next step; logits = attn_new @ Wf + bf.
__global__ __launch_bounds__(256) void k_out(const bf16_t* __restrict__ Xa,
                                             const bf16_t* __restrict__ Wa_t,
                                             bf16_t* __restrict__ Xz,
                                             const float* __restrict__ Wf,
                                             const float* __restrict__ bfv,
                                             float* __restrict__ out, int t) {
  __shared__ float an[16][256];
  int tid = threadIdx.x, lane = tid & 31, w = tid >> 5;
  int half = lane >> 4, r = lane & 15;
  int bi0 = blockIdx.x * 16;

  const bf16_t* A = Xa + (size_t)bi0 * 512;
  int n0 = (2 * w) * 16, n1 = n0 + 16;
  v8f acc0 = {}, acc1 = {};
  for (int kb = 0; kb < 512; kb += 32) {
    v16bf a  = load_frag(A, 512, r, kb, half);
    v16bf b0 = load_frag(Wa_t + (size_t)n0 * 512, 512, r, kb, half);
    v16bf b1 = load_frag(Wa_t + (size_t)n1 * 512, 512, r, kb, half);
    acc0 = wmma_bf16(a, b0, acc0);
    acc1 = wmma_bf16(a, b1, acc1);
  }
#pragma unroll
  for (int j = 0; j < 8; ++j) {
    int m = half * 8 + j;
    int b = bi0 + m;
    an[m][n0 + r] = acc0[j];
    an[m][n1 + r] = acc1[j];
    Xz[(size_t)b * 512 + n0 + r] = f2bf(acc0[j]);
    Xz[(size_t)b * 512 + n1 + r] = f2bf(acc1[j]);
  }

  // Copy h_new (Xa cols 0..255) into Xz cols 256..511 (uint = 2 bf16).
  {
    const unsigned* src = (const unsigned*)(Xa + (size_t)bi0 * 512);
    unsigned* dst = (unsigned*)(Xz + (size_t)bi0 * 512);
    for (int i = tid; i < 16 * 128; i += 256) {
      int rr = i >> 7, cc = i & 127;
      dst[rr * 256 + 128 + cc] = src[rr * 256 + cc];
    }
  }
  __syncthreads();

  if (tid < 16 * NV) {
    int bi = tid / NV, vv = tid % NV;
    float s = bfv[vv];
    for (int u = 0; u < 256; ++u) s += an[bi][u] * Wf[(size_t)u * NV + vv];
    out[(((size_t)(bi0 + bi)) * (TOUT - 1) + t) * NV + vv] = s;
  }
}

// ---------------------------------------------------------------------------

extern "C" void kernel_launch(void* const* d_in, const int* in_sizes, int n_in,
                              void* d_out, int out_size, void* d_ws, size_t ws_size,
                              hipStream_t stream) {
  const int*   ids    = (const int*)  d_in[0];
  const float* memory = (const float*)d_in[1];
  const float* enc_h  = (const float*)d_in[2];
  const float* enc_c  = (const float*)d_in[3];
  const float* Wi     = (const float*)d_in[4];
  const float* Wh     = (const float*)d_in[5];
  const float* bias   = (const float*)d_in[6];
  const float* Wm     = (const float*)d_in[7];
  const float* Wq     = (const float*)d_in[8];
  const float* vvec   = (const float*)d_in[9];
  const float* Wa     = (const float*)d_in[10];
  const float* Wf     = (const float*)d_in[11];
  const float* bfv    = (const float*)d_in[12];
  float* out = (float*)d_out;

  char* ws = (char*)d_ws;
  size_t off = 0;
  auto carve = [&](size_t bytes) {
    void* p = ws + off;
    off += (bytes + 255) & ~(size_t)255;
    return p;
  };
  bf16_t* memb  = (bf16_t*)carve((size_t)NB * TIN * NU * 2);   // 64 MiB
  bf16_t* keys  = (bf16_t*)carve((size_t)NB * TIN * NU * 2);   // 64 MiB
  bf16_t* Wzn_t = (bf16_t*)carve((size_t)1024 * 512 * 2);
  bf16_t* Wq_t  = (bf16_t*)carve((size_t)256 * 256 * 2);
  bf16_t* Wa_t  = (bf16_t*)carve((size_t)256 * 512 * 2);
  bf16_t* Wm_t  = (bf16_t*)carve((size_t)256 * 256 * 2);
  float*  zb    = (float*) carve((size_t)NV * 1024 * 4);
  bf16_t* Xz    = (bf16_t*)carve((size_t)NB * 512 * 2);        // [attn | h]
  bf16_t* Xa    = (bf16_t*)carve((size_t)NB * 512 * 2);        // [h_new | ctx]
  float*  c_st  = (float*) carve((size_t)NB * NU * 4);
  (void)ws_size; (void)in_sizes; (void)n_in; (void)out_size;

  const int n_mem = NB * TIN * NU;                       // 33554432
  k_cvt_mem<<<n_mem / 256, 256, 0, stream>>>(memory, memb, n_mem);
  k_pack<<<792576 / 256, 256, 0, stream>>>(Wi, Wh, bias, Wq, Wa, Wm,
                                           Wzn_t, Wq_t, Wa_t, Wm_t, zb);
  k_init<<<(NB * NU) / 256, 256, 0, stream>>>(enc_h, enc_c, Xz, c_st);
  k_keys<<<(NB * TIN) / 16, 256, 0, stream>>>(memb, Wm_t, keys);

  for (int t = 0; t < TOUT - 1; ++t) {
    k_lstm<<<dim3(NB / 16, NU / 16), 32, 0, stream>>>(Xz, Wzn_t, zb, ids, c_st, Xa, t);
    k_attn<<<NB / 16, 256, 0, stream>>>(Xa, Wq_t, keys, memb, vvec);
    k_out <<<NB / 16, 256, 0, stream>>>(Xa, Wa_t, Xz, Wf, bfv, out, t);
  }
}
